// MitoticTransformerBlock_78546361909751
// MI455X (gfx1250) — matrix operand order repaired and checked by reading, not compiled, not run on hardware
//
#include <hip/hip_runtime.h>

typedef _Float16 h16;
typedef __attribute__((ext_vector_type(16))) _Float16 v16h;
typedef __attribute__((ext_vector_type(8)))  _Float16 v8h;
typedef __attribute__((ext_vector_type(4)))  _Float16 v4h;
typedef __attribute__((ext_vector_type(8)))  float    v8f;
typedef __attribute__((ext_vector_type(4)))  int      v4i;

#define T_LEN   4096
#define D_DIM   1024
#define H_HEADS 16
#define HD      64
#define FF_DIM  4096
#define WIN     16
#define LN_EPS  1e-5f

// ---- CDNA5 async global->LDS path (ASYNCcnt), with sync fallback ----
#if defined(__has_builtin)
#if __has_builtin(__builtin_amdgcn_global_load_async_to_lds_b128) && \
    __has_builtin(__builtin_amdgcn_s_wait_asynccnt)
#define USE_ASYNC_LDS 1
#endif
#endif
#ifndef USE_ASYNC_LDS
#define USE_ASYNC_LDS 0
#endif

#if USE_ASYNC_LDS
typedef __attribute__((address_space(1))) v4i gv4i;   // global v4i
typedef __attribute__((address_space(3))) v4i sv4i;   // LDS v4i
__device__ __forceinline__ void async_cp_b128(void* lds, const void* g) {
  void* gg = const_cast<void*>(g);
  __builtin_amdgcn_global_load_async_to_lds_b128((gv4i*)gg, (sv4i*)lds, 0, 0);
}
__device__ __forceinline__ void async_wait0() {
  __builtin_amdgcn_s_wait_asynccnt(0);
}
#endif

// ---------------- f32 -> f16 conversion (vectorized) ----------------
__global__ void cvt_f16(const float* __restrict__ src, h16* __restrict__ dst, int n4) {
  int i = blockIdx.x * blockDim.x + threadIdx.x;
  int stride = gridDim.x * blockDim.x;
  for (int j = i; j < n4; j += stride) {
    float4 v = ((const float4*)src)[j];
    v4h o;
    o[0] = (h16)v.x; o[1] = (h16)v.y; o[2] = (h16)v.z; o[3] = (h16)v.w;
    *(v4h*)(dst + 4 * (size_t)j) = o;
  }
}

// ---------------- LayerNorm + f16 cast (one block per row) ----------------
__launch_bounds__(256)
__global__ void ln_cast(const float* __restrict__ x, const float* __restrict__ w,
                        const float* __restrict__ b, h16* __restrict__ out) {
  __shared__ float2 red[256];
  const int row = blockIdx.x;
  const int tid = threadIdx.x;
  const float* xr = x + (size_t)row * D_DIM + tid * 4;
  float xv[4];
#pragma unroll
  for (int i = 0; i < 4; ++i) xv[i] = xr[i];
  float s = xv[0] + xv[1] + xv[2] + xv[3];
  float q = xv[0]*xv[0] + xv[1]*xv[1] + xv[2]*xv[2] + xv[3]*xv[3];
  red[tid] = make_float2(s, q);
  __syncthreads();
  for (int o = 128; o > 0; o >>= 1) {
    if (tid < o) { red[tid].x += red[tid + o].x; red[tid].y += red[tid + o].y; }
    __syncthreads();
  }
  const float mu   = red[0].x * (1.f / D_DIM);
  const float var  = red[0].y * (1.f / D_DIM) - mu * mu;
  const float rstd = rsqrtf(var + LN_EPS);
  h16* orow = out + (size_t)row * D_DIM + tid * 4;
#pragma unroll
  for (int i = 0; i < 4; ++i) {
    float wv = w[tid * 4 + i], bv = b[tid * 4 + i];
    orow[i] = (h16)((xv[i] - mu) * rstd * wv + bv);
  }
}

// ---------------- Router: LN2 + 2-way sigmoid gates ----------------
__launch_bounds__(256)
__global__ void router_kn(const float* __restrict__ x2, const float* __restrict__ w,
                          const float* __restrict__ b, const float* __restrict__ rw,
                          const float* __restrict__ rb, float* __restrict__ gates) {
  __shared__ float2 red[256];
  const int row = blockIdx.x;
  const int tid = threadIdx.x;
  const float* xr = x2 + (size_t)row * D_DIM + tid * 4;
  float xv[4];
#pragma unroll
  for (int i = 0; i < 4; ++i) xv[i] = xr[i];
  float s = xv[0] + xv[1] + xv[2] + xv[3];
  float q = xv[0]*xv[0] + xv[1]*xv[1] + xv[2]*xv[2] + xv[3]*xv[3];
  red[tid] = make_float2(s, q);
  __syncthreads();
  for (int o = 128; o > 0; o >>= 1) {
    if (tid < o) { red[tid].x += red[tid + o].x; red[tid].y += red[tid + o].y; }
    __syncthreads();
  }
  const float mu   = red[0].x * (1.f / D_DIM);
  const float var  = red[0].y * (1.f / D_DIM) - mu * mu;
  const float rstd = rsqrtf(var + LN_EPS);
  __syncthreads();
  float d0 = 0.f, d1 = 0.f;
#pragma unroll
  for (int i = 0; i < 4; ++i) {
    int d = tid * 4 + i;
    float xn = (xv[i] - mu) * rstd * w[d] + b[d];
    d0 += xn * rw[d];
    d1 += xn * rw[D_DIM + d];
  }
  red[tid] = make_float2(d0, d1);
  __syncthreads();
  for (int o = 128; o > 0; o >>= 1) {
    if (tid < o) { red[tid].x += red[tid + o].x; red[tid].y += red[tid + o].y; }
    __syncthreads();
  }
  if (tid == 0) {
    gates[row * 2 + 0] = 1.f / (1.f + __expf(-(red[0].x + rb[0])));
    gates[row * 2 + 1] = 1.f / (1.f + __expf(-(red[0].y + rb[1])));
  }
}

// ---------------- Fragment helpers (ISA §7.12.2 layouts) ----------------
__device__ __forceinline__ v16h load_a_frag(const h16* As, int row, int lh) {
  const v8h lo = *(const v8h*)(As + row * 32 + 8 * lh);
  const v8h hi = *(const v8h*)(As + row * 32 + 16 + 8 * lh);
  v16h a;
#pragma unroll
  for (int e = 0; e < 8; ++e) { a[e] = lo[e]; a[e + 8] = hi[e]; }
  return a;
}
__device__ __forceinline__ v16h load_b_frag(const h16* Bs, int row, int lh) {
  const v8h lo = *(const v8h*)(Bs + row * 32 + 8 * lh);
  const v8h hi = *(const v8h*)(Bs + row * 32 + 16 + 8 * lh);
  v16h b;
#pragma unroll
  for (int e = 0; e < 8; ++e) { b[e] = lo[e]; b[e + 8] = hi[e]; }
  return b;
}

template <int NT>
__device__ __forceinline__ void mma_block(const h16* Asb, const h16* Bsb,
                                          int rA, int rB, int ll, int lh,
                                          v8f (&c)[2][NT]) {
  v16h a[2], b[NT];
#pragma unroll
  for (int mi = 0; mi < 2; ++mi) a[mi] = load_a_frag(Asb, rA + mi * 16 + ll, lh);
#pragma unroll
  for (int ni = 0; ni < NT; ++ni) b[ni] = load_b_frag(Bsb, rB + ni * 16 + ll, lh);
#pragma unroll
  for (int mi = 0; mi < 2; ++mi)
#pragma unroll
    for (int ni = 0; ni < NT; ++ni)
      c[mi][ni] = __builtin_amdgcn_wmma_f32_16x16x32_f16(
          false, a[mi], false, b[ni], (short)0, c[mi][ni], false, false);
}

// ---------------- Tiled WMMA GEMM: C[M,N] = A[M,K] x B[N,K]^T ----------------
// Block tile 128x128xBK32, double-buffered LDS, 8 waves each 32x64.
// EPI 0: outf = acc
// EPI 1: v = acc + resid; outf = v; outh = (f16)v
// EPI 2: outf += gate[row*2] * acc   (gate pre-offset by expert)
template <int EPI>
__launch_bounds__(256)
__global__ void gemm_wmma(const h16* __restrict__ A, const h16* __restrict__ B,
                          int M, int N, int K,
                          const float* __restrict__ resid,
                          float* __restrict__ outf, h16* __restrict__ outh,
                          const float* __restrict__ gate) {
  __shared__ h16 As[2][128 * 32];
  __shared__ h16 Bs[2][128 * 32];
  const int tid   = threadIdx.x;
  const int lane  = tid & 31;
  const int wid   = tid >> 5;
  const int waveM = wid >> 1;   // 0..3 -> 32-row strip
  const int waveN = wid & 1;    // 0..1 -> 64-col strip
  const int m0    = blockIdx.x * 128;
  const int n0    = blockIdx.y * 128;
  const int lh    = lane >> 4;
  const int ll    = lane & 15;

  const int sr   = tid >> 1;            // staged row 0..127
  const int sseg = (tid & 1) * 16;      // 16-half segment
  const h16* gA  = A + (size_t)(m0 + sr) * K + sseg;
  const h16* gB  = B + (size_t)(n0 + sr) * K + sseg;
  const int soff = sr * 32 + sseg;

  v8f c[2][4] = {};
  const int nk = K / 32;

#if USE_ASYNC_LDS
  async_cp_b128(&As[0][soff], gA);
  async_cp_b128(&As[0][soff + 8], gA + 8);
  async_cp_b128(&Bs[0][soff], gB);
  async_cp_b128(&Bs[0][soff + 8], gB + 8);
  async_wait0();
  __syncthreads();
  for (int k = 1; k < nk; ++k) {
    const h16* ga = gA + (size_t)k * 32;
    const h16* gb = gB + (size_t)k * 32;
    const int buf = k & 1;
    async_cp_b128(&As[buf][soff], ga);
    async_cp_b128(&As[buf][soff + 8], ga + 8);
    async_cp_b128(&Bs[buf][soff], gb);
    async_cp_b128(&Bs[buf][soff + 8], gb + 8);
    mma_block<4>(As[(k - 1) & 1], Bs[(k - 1) & 1], waveM * 32, waveN * 64, ll, lh, c);
    async_wait0();
    __syncthreads();
  }
  mma_block<4>(As[(nk - 1) & 1], Bs[(nk - 1) & 1], waveM * 32, waveN * 64, ll, lh, c);
#else
  v8h ra0 = *(const v8h*)gA, ra1 = *(const v8h*)(gA + 8);
  v8h rb0 = *(const v8h*)gB, rb1 = *(const v8h*)(gB + 8);
  *(v8h*)&As[0][soff] = ra0; *(v8h*)&As[0][soff + 8] = ra1;
  *(v8h*)&Bs[0][soff] = rb0; *(v8h*)&Bs[0][soff + 8] = rb1;
  __syncthreads();
  for (int k = 1; k < nk; ++k) {
    const h16* ga = gA + (size_t)k * 32;
    const h16* gb = gB + (size_t)k * 32;
    ra0 = *(const v8h*)ga; ra1 = *(const v8h*)(ga + 8);
    rb0 = *(const v8h*)gb; rb1 = *(const v8h*)(gb + 8);
    mma_block<4>(As[(k - 1) & 1], Bs[(k - 1) & 1], waveM * 32, waveN * 64, ll, lh, c);
    const int buf = k & 1;
    *(v8h*)&As[buf][soff] = ra0; *(v8h*)&As[buf][soff + 8] = ra1;
    *(v8h*)&Bs[buf][soff] = rb0; *(v8h*)&Bs[buf][soff + 8] = rb1;
    __syncthreads();
  }
  mma_block<4>(As[(nk - 1) & 1], Bs[(nk - 1) & 1], waveM * 32, waveN * 64, ll, lh, c);
#endif

#pragma unroll
  for (int mi = 0; mi < 2; ++mi)
#pragma unroll
    for (int ni = 0; ni < 4; ++ni)
#pragma unroll
      for (int r = 0; r < 8; ++r) {
        int row = m0 + waveM * 32 + mi * 16 + r + 8 * lh;
        int col = n0 + waveN * 64 + ni * 16 + ll;
        size_t idx = (size_t)row * N + col;
        float v = c[mi][ni][r];
        if (EPI == 0) {
          outf[idx] = v;
        } else if (EPI == 1) {
          v += resid[idx];
          outf[idx] = v;
          outh[idx] = (h16)v;
        } else {
          outf[idx] += gate[(size_t)row * 2] * v;
        }
      }
}

// ---------------- Dual-B GEMM: h = relu(A·B1^T) * (A·B2^T) -> f16 ----------------
// Block tile 128x64xBK32, double-buffered LDS, 8 waves each 32x32 (x2 accum sets).
__launch_bounds__(256)
__global__ void gemm_dual(const h16* __restrict__ A, const h16* __restrict__ B1,
                          const h16* __restrict__ B2, int M, int N, int K,
                          h16* __restrict__ outh) {
  __shared__ h16 As[2][128 * 32];
  __shared__ h16 B1s[2][64 * 32];
  __shared__ h16 B2s[2][64 * 32];
  const int tid   = threadIdx.x;
  const int lane  = tid & 31;
  const int wid   = tid >> 5;
  const int waveM = wid >> 1;
  const int waveN = wid & 1;
  const int m0    = blockIdx.x * 128;
  const int n0    = blockIdx.y * 64;
  const int lh    = lane >> 4;
  const int ll    = lane & 15;

  const int sr   = tid >> 1,  sseg = (tid & 1) * 16;   // A staging
  const int br   = tid >> 2,  bseg = (tid & 3) * 8;    // B staging (64 rows)
  const h16* gA  = A  + (size_t)(m0 + sr) * K + sseg;
  const h16* gB1 = B1 + (size_t)(n0 + br) * K + bseg;
  const h16* gB2 = B2 + (size_t)(n0 + br) * K + bseg;
  const int aoff = sr * 32 + sseg;
  const int boff = br * 32 + bseg;

  v8f c1[2][2] = {}, c2[2][2] = {};
  const int nk = K / 32;

#if USE_ASYNC_LDS
  async_cp_b128(&As[0][aoff], gA);
  async_cp_b128(&As[0][aoff + 8], gA + 8);
  async_cp_b128(&B1s[0][boff], gB1);
  async_cp_b128(&B2s[0][boff], gB2);
  async_wait0();
  __syncthreads();
  for (int k = 1; k < nk; ++k) {
    const int buf = k & 1, pb = (k - 1) & 1;
    async_cp_b128(&As[buf][aoff], gA + (size_t)k * 32);
    async_cp_b128(&As[buf][aoff + 8], gA + (size_t)k * 32 + 8);
    async_cp_b128(&B1s[buf][boff], gB1 + (size_t)k * 32);
    async_cp_b128(&B2s[buf][boff], gB2 + (size_t)k * 32);
    mma_block<2>(As[pb], B1s[pb], waveM * 32, waveN * 32, ll, lh, c1);
    mma_block<2>(As[pb], B2s[pb], waveM * 32, waveN * 32, ll, lh, c2);
    async_wait0();
    __syncthreads();
  }
  {
    const int pb = (nk - 1) & 1;
    mma_block<2>(As[pb], B1s[pb], waveM * 32, waveN * 32, ll, lh, c1);
    mma_block<2>(As[pb], B2s[pb], waveM * 32, waveN * 32, ll, lh, c2);
  }
#else
  v8h ra0 = *(const v8h*)gA, ra1 = *(const v8h*)(gA + 8);
  v8h rb1 = *(const v8h*)gB1, rb2 = *(const v8h*)gB2;
  *(v8h*)&As[0][aoff] = ra0; *(v8h*)&As[0][aoff + 8] = ra1;
  *(v8h*)&B1s[0][boff] = rb1; *(v8h*)&B2s[0][boff] = rb2;
  __syncthreads();
  for (int k = 1; k < nk; ++k) {
    ra0 = *(const v8h*)(gA + (size_t)k * 32);
    ra1 = *(const v8h*)(gA + (size_t)k * 32 + 8);
    rb1 = *(const v8h*)(gB1 + (size_t)k * 32);
    rb2 = *(const v8h*)(gB2 + (size_t)k * 32);
    const int buf = k & 1, pb = (k - 1) & 1;
    mma_block<2>(As[pb], B1s[pb], waveM * 32, waveN * 32, ll, lh, c1);
    mma_block<2>(As[pb], B2s[pb], waveM * 32, waveN * 32, ll, lh, c2);
    *(v8h*)&As[buf][aoff] = ra0; *(v8h*)&As[buf][aoff + 8] = ra1;
    *(v8h*)&B1s[buf][boff] = rb1; *(v8h*)&B2s[buf][boff] = rb2;
    __syncthreads();
  }
  {
    const int pb = (nk - 1) & 1;
    mma_block<2>(As[pb], B1s[pb], waveM * 32, waveN * 32, ll, lh, c1);
    mma_block<2>(As[pb], B2s[pb], waveM * 32, waveN * 32, ll, lh, c2);
  }
#endif

#pragma unroll
  for (int mi = 0; mi < 2; ++mi)
#pragma unroll
    for (int ni = 0; ni < 2; ++ni)
#pragma unroll
      for (int r = 0; r < 8; ++r) {
        int row = m0 + waveM * 32 + mi * 16 + r + 8 * lh;
        int col = n0 + waveN * 32 + ni * 16 + ll;
        float g = c1[mi][ni][r];
        float p = c2[mi][ni][r];
        outh[(size_t)row * N + col] = (h16)(fmaxf(g, 0.f) * p);
      }
}

// ---------------- Windowed causal attention: one wave per (q, head) ----------------
// qkv layout: [T, 3072], cols 0..1023 = Q, 1024..2047 = K, 2048..3071 = V
__launch_bounds__(256)
__global__ void attn_win(const float* __restrict__ qkv, h16* __restrict__ out) {
  const int w    = blockIdx.x * 8 + (threadIdx.x >> 5);
  const int lane = threadIdx.x & 31;
  const int q    = w / H_HEADS;
  const int h    = w % H_HEADS;
  const int RS   = 3 * D_DIM;
  const float* qp = qkv + (size_t)q * RS + h * HD;
  const float q0 = qp[lane], q1 = qp[lane + 32];

  float s[WIN];
#pragma unroll
  for (int j = 0; j < WIN; ++j) {
    const int k = q - (WIN - 1) + j;
    float part = -1e30f;
    if (k >= 0) {  // uniform across the wave
      const float* kp = qkv + (size_t)k * RS + D_DIM + h * HD;
      part = q0 * kp[lane] + q1 * kp[lane + 32];
#pragma unroll
      for (int off = 16; off >= 1; off >>= 1) part += __shfl_xor(part, off, 32);
      part *= 0.125f;  // 1/sqrt(64)
    }
    s[j] = part;
  }
  float m = s[0];
#pragma unroll
  for (int j = 1; j < WIN; ++j) m = fmaxf(m, s[j]);
  float p[WIN], sum = 0.f;
#pragma unroll
  for (int j = 0; j < WIN; ++j) { p[j] = __expf(s[j] - m); sum += p[j]; }
  const float inv = 1.f / sum;
  float o0 = 0.f, o1 = 0.f;
#pragma unroll
  for (int j = 0; j < WIN; ++j) {
    const int k = q - (WIN - 1) + j;
    if (k >= 0) {
      const float* vp = qkv + (size_t)k * RS + 2 * D_DIM + h * HD;
      o0 += p[j] * vp[lane];
      o1 += p[j] * vp[lane + 32];
    }
  }
  h16* op = out + (size_t)q * D_DIM + h * HD;
  op[lane]      = (h16)(o0 * inv);
  op[lane + 32] = (h16)(o1 * inv);
}

// ---------------- Host-side orchestration ----------------
extern "C" void kernel_launch(void* const* d_in, const int* in_sizes, int n_in,
                              void* d_out, int out_size, void* d_ws, size_t ws_size,
                              hipStream_t stream) {
  const float* x    = (const float*)d_in[0];
  const float* ln1w = (const float*)d_in[1];
  const float* ln1b = (const float*)d_in[2];
  const float* Wq   = (const float*)d_in[3];
  const float* Wk   = (const float*)d_in[4];
  const float* Wv   = (const float*)d_in[5];
  const float* Wo   = (const float*)d_in[6];
  const float* ln2w = (const float*)d_in[7];
  const float* ln2b = (const float*)d_in[8];
  const float* rw   = (const float*)d_in[9];
  const float* rb   = (const float*)d_in[10];
  const float* eg   = (const float*)d_in[11];
  const float* ep   = (const float*)d_in[12];
  const float* eo   = (const float*)d_in[13];
  float* out = (float*)d_out;

  const size_t DD = (size_t)D_DIM * D_DIM;     // 1M
  const size_t DF = (size_t)FF_DIM * D_DIM;    // 4M
  const size_t TD = (size_t)T_LEN * D_DIM;     // 4M
  const size_t TF = (size_t)T_LEN * FF_DIM;    // 16M

  char* ws = (char*)d_ws;
  size_t off = 0;
  auto alloc = [&](size_t bytes) -> void* {
    void* p = ws + off;
    off += (bytes + 255) & ~(size_t)255;
    return p;
  };
  h16*   wqkv16 = (h16*)alloc(3 * DD * 2);      // Wq|Wk|Wv rows stacked: [3072,1024]
  h16*   wo16   = (h16*)alloc(DD * 2);
  h16*   eg16   = (h16*)alloc(2 * DF * 2);
  h16*   ep16   = (h16*)alloc(2 * DF * 2);
  h16*   eo16   = (h16*)alloc(2 * DF * 2);
  h16*   xn16   = (h16*)alloc(TD * 2);
  float* qkvf   = (float*)alloc(3 * TD * 4);    // [T, 3072] f32
  h16*   attn16 = (h16*)alloc(TD * 2);
  h16*   x2f16  = (h16*)alloc(TD * 2);
  float* gates  = (float*)alloc((size_t)T_LEN * 2 * 4);
  h16*   hbuf   = (h16*)alloc(TF * 2);

  // 1) weights -> f16 (f16 weight set ~56MB: L2-resident on 192MB L2)
  cvt_f16<<<1024, 256, 0, stream>>>(Wq, wqkv16,          (int)(DD / 4));
  cvt_f16<<<1024, 256, 0, stream>>>(Wk, wqkv16 + DD,     (int)(DD / 4));
  cvt_f16<<<1024, 256, 0, stream>>>(Wv, wqkv16 + 2 * DD, (int)(DD / 4));
  cvt_f16<<<1024, 256, 0, stream>>>(Wo, wo16,            (int)(DD / 4));
  cvt_f16<<<2048, 256, 0, stream>>>(eg, eg16, (int)(2 * DF / 4));
  cvt_f16<<<2048, 256, 0, stream>>>(ep, ep16, (int)(2 * DF / 4));
  cvt_f16<<<2048, 256, 0, stream>>>(eo, eo16, (int)(2 * DF / 4));

  // 2) LN1 + cast
  ln_cast<<<T_LEN, 256, 0, stream>>>(x, ln1w, ln1b, xn16);

  // 3) fused QKV projection: [T,1024] x [3072,1024]^T -> [T,3072]
  gemm_wmma<0><<<dim3(T_LEN / 128, 3 * D_DIM / 128), 256, 0, stream>>>(
      xn16, wqkv16, T_LEN, 3 * D_DIM, D_DIM, nullptr, qkvf, nullptr, nullptr);

  // 4) windowed attention
  attn_win<<<(T_LEN * H_HEADS) / 8, 256, 0, stream>>>(qkvf, attn16);

  // 5) output projection + residual -> d_out (f32) and x2 (f16)
  dim3 gD(T_LEN / 128, D_DIM / 128);
  gemm_wmma<1><<<gD, 256, 0, stream>>>(attn16, wo16, T_LEN, D_DIM, D_DIM,
                                       x, out, x2f16, nullptr);

  // 6) router gates from LN2(x2)
  router_kn<<<T_LEN, 256, 0, stream>>>(out, ln2w, ln2b, rw, rb, gates);

  // 7) experts: h = relu(x2·eg^T)*(x2·ep^T); out += gate_i * (h·eo^T)
  dim3 gFF(T_LEN / 128, FF_DIM / 64);
  for (int i = 0; i < 2; ++i) {
    gemm_dual<<<gFF, 256, 0, stream>>>(x2f16, eg16 + (size_t)i * DF, ep16 + (size_t)i * DF,
                                       T_LEN, FF_DIM, D_DIM, hbuf);
    gemm_wmma<2><<<gD, 256, 0, stream>>>(hbuf, eo16 + (size_t)i * DF, T_LEN, D_DIM, FF_DIM,
                                         nullptr, out, nullptr, gates + i);
  }
}
